// ResLayer_558345749246
// MI455X (gfx1250) — compile-verified, matrix operand order checked by reading, not run on hardware
//
#include <hip/hip_runtime.h>
#include <hip/hip_bf16.h>
#include <math.h>

#define HEADS     4
#define DIMS      64
#define FEAT      256
#define EDGE_DIM  16
#define NEG_SLOPE 0.2f

typedef __bf16 bf16_t;
typedef __attribute__((ext_vector_type(16))) __bf16 v16bf;
typedef __attribute__((ext_vector_type(8)))  __bf16 v8bf;
typedef __attribute__((ext_vector_type(8)))  float  v8f;
typedef __attribute__((ext_vector_type(4)))  int    v4i;

union bfrag { v16bf v; struct { v8bf lo, hi; } h; };

// gfx1250 async global->LDS staging (ASYNCcnt path), with compile-safe fallback
#if __has_builtin(__builtin_amdgcn_global_load_async_to_lds_b128) && \
    __has_builtin(__builtin_amdgcn_s_wait_asynccnt)
#define USE_ASYNC_LDS 1
#else
#define USE_ASYNC_LDS 0
#endif

// ---------------------------------------------------------------------------
// fp32 -> monotonic unsigned mapping for atomicMax-based segment max
// ---------------------------------------------------------------------------
__device__ __forceinline__ unsigned f2ord(float f) {
    unsigned b = __float_as_uint(f);
    return (b & 0x80000000u) ? ~b : (b | 0x80000000u);
}
__device__ __forceinline__ float ord2f(unsigned o) {
    unsigned b = (o & 0x80000000u) ? (o & 0x7FFFFFFFu) : ~o;
    return __uint_as_float(b);
}

// ---------------------------------------------------------------------------
// Convert both weight matrices fp32 -> bf16 (tiny: 2 x 256x256)
// ---------------------------------------------------------------------------
__global__ void cvt_w_kernel(const float* __restrict__ a, const float* __restrict__ b,
                             bf16_t* __restrict__ oa, bf16_t* __restrict__ ob, int n) {
    int i = blockIdx.x * blockDim.x + threadIdx.x;
    if (i < n) { oa[i] = (bf16_t)a[i]; ob[i] = (bf16_t)b[i]; }
}

// x fp32 -> bf16 (so GEMM tiles can be staged with async b128 copies)
__global__ void cvt_x_kernel(const float* __restrict__ x, bf16_t* __restrict__ xb, int n) {
    int i = blockIdx.x * blockDim.x + threadIdx.x;
    if (i < n) xb[i] = (bf16_t)x[i];
}

// ---------------------------------------------------------------------------
// Zero / init scratch buffers (deg, loop_sum, amax, denom, out_acc)
// ---------------------------------------------------------------------------
__global__ void init_kernel(float* __restrict__ deg, float* __restrict__ loop_sum,
                            unsigned* __restrict__ amax, float* __restrict__ denom,
                            float* __restrict__ out_acc, int N) {
    int i = blockIdx.x * blockDim.x + threadIdx.x;
    if (i < N) deg[i] = 0.0f;
    if (i < N * EDGE_DIM) loop_sum[i] = 0.0f;
    if (i < N * HEADS) { amax[i] = 0u; denom[i] = 0.0f; }
    if (i < N * FEAT) out_acc[i] = 0.0f;
}

// ---------------------------------------------------------------------------
// Fused: xl = x@W_l^T + b_l, xr = x@W_r^T + b_r  via v_wmma_f32_16x16x32_bf16
// Block: 256 threads (8 waves), M-tile = 64 rows of x (bf16, 32KB LDS),
// staged once (async global->LDS when available) and reused for BOTH weight
// matrices. Each wave owns 2 N-tiles; each B fragment is reused across the
// 4 M-subtiles -> 8 WMMAs per 32-wide k-step per wave.
// ---------------------------------------------------------------------------
__global__ __launch_bounds__(256) void node_gemm_kernel(
    const bf16_t* __restrict__ xb,    // [N,256] bf16
    const bf16_t* __restrict__ Wlb,   // [256,256] bf16 row-major
    const bf16_t* __restrict__ Wrb,
    const float*  __restrict__ b_l,   // [256]
    const float*  __restrict__ b_r,
    float*        __restrict__ xl,    // [N,256] fp32
    float*        __restrict__ xr,
    int N)
{
    __shared__ __attribute__((aligned(16))) bf16_t axs[64 * FEAT];  // 32 KB

    const int row0 = blockIdx.x * 64;
    const int tid  = threadIdx.x;
    const bool full = (row0 + 64 <= N);

    // ---- stage 64x256 bf16 x-tile into LDS (contiguous 32KB region) ----
    if (full) {
#if USE_ASYNC_LDS
        const bf16_t* gsrc = xb + (size_t)row0 * FEAT;
        #pragma unroll
        for (int i = 0; i < 8; ++i) {
            int off = (tid + i * 256) * 8;  // 8 bf16 = 16B per chunk
            __builtin_amdgcn_global_load_async_to_lds_b128(
                (__attribute__((address_space(1))) v4i*)(gsrc + off),
                (__attribute__((address_space(3))) v4i*)(axs + off),
                0, 0);
        }
        __builtin_amdgcn_s_wait_asynccnt(0);
#else
        const bf16_t* gsrc = xb + (size_t)row0 * FEAT;
        #pragma unroll
        for (int i = 0; i < 8; ++i) {
            int off = (tid + i * 256) * 8;
            *(v8bf*)&axs[off] = *(const v8bf*)&gsrc[off];
        }
#endif
    } else {
        for (int c = tid; c < 64 * FEAT / 8; c += 256) {
            int off = c * 8;
            int gr  = row0 + (off >> 8);
            v8bf v  = {};
            if (gr < N) v = *(const v8bf*)&xb[(size_t)gr * FEAT + (off & 255)];
            *(v8bf*)&axs[off] = v;
        }
    }
    __syncthreads();

    const int wave  = tid >> 5;
    const int lane  = tid & 31;
    const int mloc  = lane & 15;          // A row / B col within a 16-tile
    const int khalf = (lane >> 4) * 8;    // lanes 16-31 hold the +8 K half
    const int mtop  = (lane >> 4) * 8;    // D rows: +8 for upper lane half
    const int col0  = wave * 32;          // 2 N-tiles per wave

    for (int wsel = 0; wsel < 2; ++wsel) {
        const bf16_t* Wb   = wsel ? Wrb : Wlb;
        const float*  bvec = wsel ? b_r : b_l;
        float*        out  = wsel ? xr  : xl;

        v8f acc[2][4] = {};
        for (int kk = 0; kk < FEAT; kk += 32) {
            bfrag afr[4];
            #pragma unroll
            for (int m = 0; m < 4; ++m) {
                const bf16_t* ap = &axs[(m * 16 + mloc) * FEAT + kk + khalf];
                afr[m].h.lo = *(const v8bf*)ap;
                afr[m].h.hi = *(const v8bf*)(ap + 16);
            }
            #pragma unroll
            for (int t = 0; t < 2; ++t) {
                const bf16_t* wrow = Wb + (size_t)(col0 + t * 16 + mloc) * FEAT + kk + khalf;
                bfrag b;
                b.h.lo = *(const v8bf*)wrow;
                b.h.hi = *(const v8bf*)(wrow + 16);
                #pragma unroll
                for (int m = 0; m < 4; ++m)
                    acc[t][m] = __builtin_amdgcn_wmma_f32_16x16x32_bf16(
                        false, afr[m].v, false, b.v, (short)0, acc[t][m], false, false);
            }
        }

        // D: f32 16x16, VGPR r -> row r (+8 for lanes 16-31), col = lane&15
        if (full) {
            #pragma unroll
            for (int t = 0; t < 2; ++t) {
                const int col  = col0 + t * 16 + mloc;
                const float bb = bvec[col];
                #pragma unroll
                for (int m = 0; m < 4; ++m) {
                    #pragma unroll
                    for (int r = 0; r < 8; ++r)
                        out[(size_t)(row0 + m * 16 + mtop + r) * FEAT + col] = acc[t][m][r] + bb;
                }
            }
        } else {
            #pragma unroll
            for (int t = 0; t < 2; ++t) {
                const int col  = col0 + t * 16 + mloc;
                const float bb = bvec[col];
                #pragma unroll
                for (int m = 0; m < 4; ++m) {
                    #pragma unroll
                    for (int r = 0; r < 8; ++r) {
                        int grow = row0 + m * 16 + mtop + r;
                        if (grow < N) out[(size_t)grow * FEAT + col] = acc[t][m][r] + bb;
                    }
                }
            }
        }
    }
}

// ---------------------------------------------------------------------------
// In-degree + loop_attr sums: one thread per (edge, attr-component)
// ---------------------------------------------------------------------------
__global__ void deg_kernel(const int* __restrict__ dst, const float* __restrict__ edge_attr,
                           float* __restrict__ deg, float* __restrict__ loop_sum, int E) {
    int idx = blockIdx.x * blockDim.x + threadIdx.x;
    if (idx >= E * EDGE_DIM) return;
    int e = idx >> 4, j = idx & 15;
    int d = dst[e];
    atomicAdd(&loop_sum[(size_t)d * EDGE_DIM + j], edge_attr[idx]);
    if (j == 0) atomicAdd(&deg[d], 1.0f);
}

// ---------------------------------------------------------------------------
// Per-edge GATv2 score: alpha = leaky_relu(x_l[src]+x_r[dst]+W_e e) . att
// One wave per edge (grid-stride). 8 fp32 features per lane; lanes 8k..8k+7
// cover one head -> 8-lane shuffle reduction. Segment max via atomicMax.
// Self-loop edges (e >= E) use mean-filled attrs computed from deg/loop_sum.
// ---------------------------------------------------------------------------
__global__ __launch_bounds__(256) void alpha_kernel(
    const int* __restrict__ src, const int* __restrict__ dst,
    const float* __restrict__ edge_attr,
    const float* __restrict__ W_e,    // [256,16]
    const float* __restrict__ att,    // [4*64] = [256], head-major
    const float* __restrict__ xl, const float* __restrict__ xr,
    const float* __restrict__ deg, const float* __restrict__ loop_sum,
    float* __restrict__ alpha, unsigned* __restrict__ amax,
    int E, int N)
{
    __shared__ float We_s[FEAT * EDGE_DIM];   // 16 KB
    __shared__ float att_s[FEAT];
    for (int i = threadIdx.x; i < FEAT * EDGE_DIM; i += blockDim.x) We_s[i] = W_e[i];
    for (int i = threadIdx.x; i < FEAT; i += blockDim.x) att_s[i] = att[i];
    __syncthreads();

    const int lane   = threadIdx.x & 31;
    const int wgid   = (blockIdx.x * blockDim.x + threadIdx.x) >> 5;
    const int nwaves = (gridDim.x * blockDim.x) >> 5;
    const int Etot   = E + N;
    const int head   = lane >> 3;
    const int f0     = lane * 8;

    for (int e = wgid; e < Etot; e += nwaves) {
        int s, d;
        float aj;
        if (e < E) {
            s = src[e]; d = dst[e];
            aj = (lane < EDGE_DIM) ? edge_attr[(size_t)e * EDGE_DIM + lane] : 0.0f;
        } else {
            s = d = e - E;
            float inv = 1.0f / fmaxf(deg[s], 1.0f);
            aj = (lane < EDGE_DIM) ? loop_sum[(size_t)s * EDGE_DIM + lane] * inv : 0.0f;
        }
        float attr[EDGE_DIM];
        #pragma unroll
        for (int j = 0; j < EDGE_DIM; ++j) attr[j] = __shfl(aj, j, 32);

        const float* xls = xl + (size_t)s * FEAT;
        const float* xrd = xr + (size_t)d * FEAT;
        float partial = 0.0f;
        #pragma unroll
        for (int j = 0; j < 8; ++j) {
            int f = f0 + j;
            float ef = 0.0f;
            const float* wrow = &We_s[f * EDGE_DIM];
            #pragma unroll
            for (int k2 = 0; k2 < EDGE_DIM; ++k2) ef = fmaf(attr[k2], wrow[k2], ef);
            float sv = xls[f] + xrd[f] + ef;
            sv = (sv > 0.0f) ? sv : NEG_SLOPE * sv;
            partial = fmaf(sv, att_s[f], partial);
        }
        partial += __shfl_xor(partial, 1, 32);
        partial += __shfl_xor(partial, 2, 32);
        partial += __shfl_xor(partial, 4, 32);
        if ((lane & 7) == 0) {
            alpha[(size_t)e * HEADS + head] = partial;
            atomicMax(&amax[(size_t)d * HEADS + head], f2ord(partial));
        }
    }
}

// ---------------------------------------------------------------------------
// ex = exp(alpha - amax[dst]); denom[dst] += ex. One thread per (edge, head).
// ---------------------------------------------------------------------------
__global__ void softmax_kernel(const int* __restrict__ dst, const unsigned* __restrict__ amax,
                               float* __restrict__ alpha, float* __restrict__ denom,
                               int E, int N) {
    int idx = blockIdx.x * blockDim.x + threadIdx.x;
    if (idx >= (E + N) * HEADS) return;
    int e = idx >> 2, h = idx & 3;
    int d = (e < E) ? dst[e] : (e - E);
    float m  = ord2f(amax[(size_t)d * HEADS + h]);
    float ex = __expf(alpha[idx] - m);
    alpha[idx] = ex;
    atomicAdd(&denom[(size_t)d * HEADS + h], ex);
}

// ---------------------------------------------------------------------------
// out[dst] += (ex/denom) * x_l[src]. One wave per edge, 8 features per lane.
// ---------------------------------------------------------------------------
__global__ __launch_bounds__(256) void aggregate_kernel(
    const int* __restrict__ src, const int* __restrict__ dst,
    const float* __restrict__ alpha, const float* __restrict__ denom,
    const float* __restrict__ xl, float* __restrict__ out_acc, int E, int N)
{
    const int lane   = threadIdx.x & 31;
    const int wgid   = (blockIdx.x * blockDim.x + threadIdx.x) >> 5;
    const int nwaves = (gridDim.x * blockDim.x) >> 5;
    const int Etot   = E + N;
    const int head   = lane >> 3;
    const int f0     = lane * 8;

    for (int e = wgid; e < Etot; e += nwaves) {
        int s, d;
        if (e < E) { s = src[e]; d = dst[e]; } else { s = d = e - E; }
        float ex = alpha[(size_t)e * HEADS + head];
        float dn = denom[(size_t)d * HEADS + head];
        float w  = ex / (dn + 1e-16f);
        const float* xls = xl + (size_t)s * FEAT;
        float* od        = out_acc + (size_t)d * FEAT;
        #pragma unroll
        for (int j = 0; j < 8; ++j)
            atomicAdd(&od[f0 + j], w * xls[f0 + j]);
    }
}

// ---------------------------------------------------------------------------
// out = x + silu(out_acc + bias)
// ---------------------------------------------------------------------------
__global__ void finalize_kernel(const float* __restrict__ x, const float* __restrict__ out_acc,
                                const float* __restrict__ bias, float* __restrict__ out,
                                int total) {
    int i = blockIdx.x * blockDim.x + threadIdx.x;
    if (i >= total) return;
    float h   = out_acc[i] + bias[i & (FEAT - 1)];
    float sig = 1.0f / (1.0f + __expf(-h));
    out[i] = x[i] + h * sig;
}

// ---------------------------------------------------------------------------
extern "C" void kernel_launch(void* const* d_in, const int* in_sizes, int n_in,
                              void* d_out, int out_size, void* d_ws, size_t ws_size,
                              hipStream_t stream)
{
    const float* x         = (const float*)d_in[0];
    const int*   eidx      = (const int*)d_in[1];
    const float* edge_attr = (const float*)d_in[2];
    const float* W_l       = (const float*)d_in[3];
    const float* b_l       = (const float*)d_in[4];
    const float* W_r       = (const float*)d_in[5];
    const float* b_r       = (const float*)d_in[6];
    const float* W_e       = (const float*)d_in[7];
    const float* att       = (const float*)d_in[8];
    const float* bias      = (const float*)d_in[9];
    float* out = (float*)d_out;

    const int N = in_sizes[0] / FEAT;
    const int E = in_sizes[1] / 2;
    const int* src = eidx;
    const int* dst = eidx + E;

    // carve workspace (~192 MB total)
    char* p = (char*)d_ws;
    auto carve = [&](size_t bytes) {
        char* r = p;
        p += (bytes + 255) & ~(size_t)255;
        return (void*)r;
    };
    float*    xl       = (float*)carve((size_t)N * FEAT * sizeof(float));
    float*    xr       = (float*)carve((size_t)N * FEAT * sizeof(float));
    float*    out_acc  = (float*)carve((size_t)N * FEAT * sizeof(float));
    bf16_t*   xb       = (bf16_t*)carve((size_t)N * FEAT * sizeof(bf16_t));
    bf16_t*   Wlb      = (bf16_t*)carve((size_t)FEAT * FEAT * sizeof(bf16_t));
    bf16_t*   Wrb      = (bf16_t*)carve((size_t)FEAT * FEAT * sizeof(bf16_t));
    float*    degb     = (float*)carve((size_t)N * sizeof(float));
    float*    loop_sum = (float*)carve((size_t)N * EDGE_DIM * sizeof(float));
    unsigned* amax     = (unsigned*)carve((size_t)N * HEADS * sizeof(unsigned));
    float*    denom    = (float*)carve((size_t)N * HEADS * sizeof(float));
    float*    alpha    = (float*)carve((size_t)(E + N) * HEADS * sizeof(float));

    // 1. weights + x -> bf16
    cvt_w_kernel<<<(FEAT * FEAT + 255) / 256, 256, 0, stream>>>(W_l, W_r, Wlb, Wrb, FEAT * FEAT);
    cvt_x_kernel<<<(N * FEAT + 255) / 256, 256, 0, stream>>>(x, xb, N * FEAT);

    // 2. init scratch
    init_kernel<<<(N * FEAT + 255) / 256, 256, 0, stream>>>(degb, loop_sum, amax, denom, out_acc, N);

    // 3. fused node GEMMs (WMMA bf16 -> f32)
    node_gemm_kernel<<<(N + 63) / 64, 256, 0, stream>>>(xb, Wlb, Wrb, b_l, b_r, xl, xr, N);

    // 4. degree + loop attr sums
    deg_kernel<<<(E * EDGE_DIM + 255) / 256, 256, 0, stream>>>(dst, edge_attr, degb, loop_sum, E);

    // 5. per-edge attention scores + segment max
    alpha_kernel<<<4096, 256, 0, stream>>>(src, dst, edge_attr, W_e, att, xl, xr,
                                           degb, loop_sum, alpha, amax, E, N);

    // 6. exp + denominator
    softmax_kernel<<<((E + N) * HEADS + 255) / 256, 256, 0, stream>>>(dst, amax, alpha, denom, E, N);

    // 7. weighted aggregation
    aggregate_kernel<<<4096, 256, 0, stream>>>(src, dst, alpha, denom, xl, out_acc, E, N);

    // 8. SiLU + residual
    finalize_kernel<<<(N * FEAT + 255) / 256, 256, 0, stream>>>(x, out_acc, bias, out, N * FEAT);
}
